// FoundationTransformerLayer_29429115912814
// MI455X (gfx1250) — matrix operand order repaired and checked by reading, not compiled, not run on hardware
//
#include <hip/hip_runtime.h>

// ---------------------------------------------------------------------------
// FoundationTransformerLayer for MI455X (gfx1250, wave32, WMMA bf16 + TDM)
// B=4 S=1024 D=512 H=8 dk=64 R=4 FF=2048 C=128 SCALES={1,4,16,64}
// ---------------------------------------------------------------------------

typedef __attribute__((ext_vector_type(16))) __bf16 v16bf;
typedef __attribute__((ext_vector_type(8)))  __bf16 v8bf;
typedef __attribute__((ext_vector_type(8)))  float  v8f;
typedef __attribute__((ext_vector_type(4)))  unsigned int v4u;
typedef __attribute__((ext_vector_type(8)))  int v8i;
typedef __attribute__((ext_vector_type(4)))  int v4i;

#define B_  4
#define S_  1024
#define D_  512
#define H_  8
#define DK_ 64
#define FF_ 2048
#define C_  128
#define M_  (B_ * S_)   // 4096 rows

// ---------------------------------------------------------------------------
// Fragment loaders (16-bit WMMA operands, contiguous k: element s of lane half hi is k0 + 16*hi + s)
// ---------------------------------------------------------------------------
__device__ __forceinline__ v16bf load_a_frag(const __bf16* __restrict__ A,
                                             size_t ld, int row_base, int k0,
                                             int l, int hi) {
    const __bf16* p = A + (size_t)(row_base + l) * ld + (size_t)k0 + hi * 16;
    v8bf lo = *(const v8bf*)(p);
    v8bf up = *(const v8bf*)(p + 8);
    v16bf r;
#pragma unroll
    for (int i = 0; i < 8; ++i) { r[i] = lo[i]; r[i + 8] = up[i]; }
    return r;
}

__device__ __forceinline__ v16bf load_b_frag(const __bf16* __restrict__ W,
                                             size_t ld, int col_base, int k0,
                                             int l, int hi) {
    const __bf16* p = W + (size_t)(col_base + l) * ld + (size_t)k0 + hi * 16;
    return *(const v16bf*)p;   // 32B aligned -> two b128 loads
}

// TDM: DMA a 64(rows) x 64(cols) bf16 tile from a row-major [*,512] tensor
// into LDS at byte offset lds_off.  D# per ISA 8.3/8.4 (2-D, groups 2/3 zero).
__device__ __forceinline__ void tdm_load_tile64(unsigned int lds_off,
                                                const __bf16* gptr) {
    const unsigned long long ga = (unsigned long long)(uintptr_t)gptr;
    v4u g0;
    g0[0] = 1u;                                // count=1, user descriptor
    g0[1] = lds_off;                           // lds_addr
    g0[2] = (unsigned int)ga;                  // global_addr[31:0]
    g0[3] = ((unsigned int)(ga >> 32) & 0x01FFFFFFu) | 0x80000000u; // addr[56:32] | type=2
    v8i g1;
    g1[0] = 0x10000;                 // data_size=1 (2 bytes)
    g1[1] = (int)(512u << 16);       // tensor_dim0 = 512      (bits 79:48)
    g1[2] = (int)(4096u << 16);      // tensor_dim1 = 4096     (bits 111:80)
    g1[3] = (int)(64u << 16);        // tile_dim0 = 64         (bits 127:112)
    g1[4] = 64;                      // tile_dim1 = 64         (bits 143:128)
    g1[5] = 512;                     // tensor_dim0_stride = 512
    g1[6] = 0;
    g1[7] = 0;
    v4i gz4 = {0, 0, 0, 0};
    v8i gz8 = {0, 0, 0, 0, 0, 0, 0, 0};
    __builtin_amdgcn_tensor_load_to_lds(g0, g1, gz4, gz4, gz8, 0);
}

// ---------------------------------------------------------------------------
// Generic bf16 WMMA GEMM:  out[M,N] = act( A[M,K] * W[N,K]^T + bias[N] )
// One wave computes a 64x64 tile: 4x4 v8f accumulators, K stepped by 32.
// Prefetches the next K-tile via global_prefetch_b8.
// ---------------------------------------------------------------------------
template <int RELU>
__global__ __launch_bounds__(256)
void gemm_bf16_kernel(const __bf16* __restrict__ A,
                      const __bf16* __restrict__ W,
                      const float* __restrict__ bias,
                      float* __restrict__ outF,
                      __bf16* __restrict__ outB,
                      int M, int N, int K) {
    const int tid  = threadIdx.x;
    const int lane = tid & 31;
    const int wv   = tid >> 5;
    const int l    = lane & 15;
    const int hi   = lane >> 4;
    const int Nt   = N >> 6;
    const int wid  = blockIdx.x * 8 + wv;
    const int mrow = (wid / Nt) * 64;
    const int ncol = (wid % Nt) * 64;

    v8f acc[4][4] = {};
    for (int k0 = 0; k0 < K; k0 += 32) {
        if (k0 + 32 < K) {   // prefetch next K-tile (global_prefetch_b8)
            __builtin_prefetch((const void*)(A + (size_t)(mrow + l) * K + k0 + 32), 0, 1);
            __builtin_prefetch((const void*)(W + (size_t)(ncol + l) * K + k0 + 32), 0, 1);
        }
        v16bf a[4], b[4];
#pragma unroll
        for (int mi = 0; mi < 4; ++mi)
            a[mi] = load_a_frag(A, (size_t)K, mrow + mi * 16, k0, l, hi);
#pragma unroll
        for (int ni = 0; ni < 4; ++ni)
            b[ni] = load_b_frag(W, (size_t)K, ncol + ni * 16, k0, l, hi);
#pragma unroll
        for (int mi = 0; mi < 4; ++mi)
#pragma unroll
            for (int ni = 0; ni < 4; ++ni)
                acc[mi][ni] = __builtin_amdgcn_wmma_f32_16x16x32_bf16(
                    false, a[mi], false, b[ni], (short)0, acc[mi][ni],
                    false, false);
    }

#pragma unroll
    for (int mi = 0; mi < 4; ++mi) {
#pragma unroll
        for (int ni = 0; ni < 4; ++ni) {
            const int col = ncol + ni * 16 + l;
            const float bv = bias ? bias[col] : 0.0f;
#pragma unroll
            for (int r = 0; r < 8; ++r) {
                const int row = mrow + mi * 16 + hi * 8 + r;
                float v = acc[mi][ni][r] + bv;
                if (RELU) v = fmaxf(v, 0.0f);
                const size_t o = (size_t)row * N + col;
                if (outF) outF[o] = v;
                if (outB) outB[o] = (__bf16)v;
            }
        }
    }
}

// ---------------------------------------------------------------------------
// Flash attention with resource-bias.  Grid: B*H*(S/64) blocks of 128 threads
// (4 waves).  Each wave owns 16 Q rows; KV tiles of 64.  dk=64.
// V tiles are DMA'd into LDS by the Tensor Data Mover, double-buffered:
// tile t+1's DMA is issued before tile t is consumed, then s_wait_tensorcnt 1
// guarantees tile t completion (TDM ops of one wave complete in order).
//
// Dynamic LDS layout:
//   [0,      8192) : V tile buffer 0 (64x64 bf16)
//   [8192,  16384) : V tile buffer 1 (64x64 bf16)
//   [16384, 24576) : per-wave P tiles, 4 x (16x64) bf16
//   [24576, 24640) : rbias slice (16 floats)
// ---------------------------------------------------------------------------
#define ATT_SMEM_BYTES 24640

__global__ __launch_bounds__(128)
void flash_attn_kernel(const __bf16* __restrict__ qb,
                       const __bf16* __restrict__ kb,
                       const __bf16* __restrict__ vb,
                       const float* __restrict__ rbias,
                       const int* __restrict__ rm,
                       const int* __restrict__ am,
                       __bf16* __restrict__ ob) {
    extern __shared__ char smem[];
    __bf16* vt0  = (__bf16*)smem;                  // V tile ping
    __bf16* vt1  = (__bf16*)(smem + 8192);         // V tile pong
    __bf16* pbuf = (__bf16*)(smem + 16384);        // 4 waves x 16x64
    float*  lrb  = (float*)(smem + 24576);         // rbias slice

    const int tid  = threadIdx.x;
    const int lane = tid & 31;
    const int wv   = tid >> 5;
    const int l    = lane & 15;
    const int hi   = lane >> 4;

    const int nQB  = S_ / 64;
    const int qblk = blockIdx.x % nQB;
    const int bh   = blockIdx.x / nQB;
    const int b    = bh / H_;
    const int h    = bh % H_;

    if (tid < 16) lrb[tid] = rbias[h * 16 + tid];

    const int    qbase  = qblk * 64 + wv * 16;
    const size_t rowoff = (size_t)b * S_;
    const __bf16* qbase_p = qb + rowoff * D_ + h * DK_;
    const __bf16* kbase_p = kb + rowoff * D_ + h * DK_;
    const __bf16* vbase_p = vb + rowoff * D_ + h * DK_;

    // LDS byte offsets (flat addr low 32 bits == LDS offset)
    const unsigned int vt_lds[2] = {
        (unsigned int)(uintptr_t)vt0, (unsigned int)(uintptr_t)vt1 };

    v16bf qf[2];
#pragma unroll
    for (int kc = 0; kc < 2; ++kc)
        qf[kc] = load_a_frag(qbase_p, (size_t)D_, qbase, kc * 32, l, hi);

    int rmi[8];
#pragma unroll
    for (int r = 0; r < 8; ++r)
        rmi[r] = rm[rowoff + qbase + hi * 8 + r];

    float m[8], lsum[8];
#pragma unroll
    for (int r = 0; r < 8; ++r) { m[r] = -3.0e38f; lsum[r] = 0.0f; }
    v8f oacc[4] = {};

    __bf16* pw = pbuf + wv * (16 * 64);

    // Prologue: DMA first V tile into buffer 0.
    if (wv == 0) tdm_load_tile64(vt_lds[0], vbase_p);
    __syncthreads();   // lrb written; TDM in flight

    const int nT = S_ / 64;
    for (int t = 0; t < nT; ++t) {
        const int j0 = t * 64;
        // ---- TDM ping-pong: launch t+1, then ensure tile t has landed ----
        if (wv == 0) {
            if (t + 1 < nT) {
                tdm_load_tile64(vt_lds[(t + 1) & 1],
                                vbase_p + (size_t)(j0 + 64) * D_);
                __builtin_amdgcn_s_wait_tensorcnt(1);   // tile t complete
            } else {
                __builtin_amdgcn_s_wait_tensorcnt(0);
            }
        }
        __syncthreads();   // V tile t visible to all 4 waves
        const __bf16* vt = (t & 1) ? vt1 : vt0;

        // ---- preload all K fragments for this tile ----
        v16bf kf[4][2];
        int   rmjv[4];
        float mkv[4];
#pragma unroll
        for (int ni = 0; ni < 4; ++ni) {
            const int col = j0 + ni * 16 + l;
            const __bf16* kp = kbase_p + (size_t)col * D_ + hi * 16;
            kf[ni][0] = *(const v16bf*)(kp);
            kf[ni][1] = *(const v16bf*)(kp + 32);
            rmjv[ni]  = rm[rowoff + col];
            mkv[ni]   = (am[rowoff + col] == 0) ? -1.0e9f : 0.0f;
        }
        // ---- scores tile: S = Q * K^T / sqrt(dk) + rbias + mask ----
        float p[4][8];
#pragma unroll
        for (int ni = 0; ni < 4; ++ni) {
            v8f s = {};
            s = __builtin_amdgcn_wmma_f32_16x16x32_bf16(
                false, qf[0], false, kf[ni][0], (short)0, s, false, false);
            s = __builtin_amdgcn_wmma_f32_16x16x32_bf16(
                false, qf[1], false, kf[ni][1], (short)0, s, false, false);
#pragma unroll
            for (int r = 0; r < 8; ++r)
                p[ni][r] = s[r] * 0.125f + lrb[rmi[r] * 4 + rmjv[ni]] + mkv[ni];
        }
        // ---- online softmax (row reductions over 16-lane groups) ----
#pragma unroll
        for (int r = 0; r < 8; ++r) {
            float mx = fmaxf(fmaxf(p[0][r], p[1][r]), fmaxf(p[2][r], p[3][r]));
            mx = fmaxf(mx, __shfl_xor(mx, 1, 32));
            mx = fmaxf(mx, __shfl_xor(mx, 2, 32));
            mx = fmaxf(mx, __shfl_xor(mx, 4, 32));
            mx = fmaxf(mx, __shfl_xor(mx, 8, 32));
            const float mn = fmaxf(m[r], mx);
            const float sc = __expf(m[r] - mn);
            m[r] = mn;
            float rs = 0.0f;
#pragma unroll
            for (int ni = 0; ni < 4; ++ni) {
                p[ni][r] = __expf(p[ni][r] - mn);
                rs += p[ni][r];
            }
            rs += __shfl_xor(rs, 1, 32);
            rs += __shfl_xor(rs, 2, 32);
            rs += __shfl_xor(rs, 4, 32);
            rs += __shfl_xor(rs, 8, 32);
            lsum[r] = lsum[r] * sc + rs;
#pragma unroll
            for (int ni = 0; ni < 4; ++ni) oacc[ni][r] *= sc;
        }
        // ---- relayout P: C-layout -> A-layout via per-wave LDS ----
#pragma unroll
        for (int ni = 0; ni < 4; ++ni)
#pragma unroll
            for (int r = 0; r < 8; ++r)
                pw[(hi * 8 + r) * 64 + ni * 16 + l] = (__bf16)p[ni][r];
        __syncthreads();
        v16bf pA[2];
#pragma unroll
        for (int kc = 0; kc < 2; ++kc) {
            const __bf16* pp = pw + l * 64 + kc * 32 + hi * 16;
            v8bf lo = *(const v8bf*)(pp);
            v8bf up = *(const v8bf*)(pp + 8);
            v16bf tt;
#pragma unroll
            for (int i = 0; i < 8; ++i) { tt[i] = lo[i]; tt[i + 8] = up[i]; }
            pA[kc] = tt;
        }
        // ---- preload all V fragments from the TDM-staged tile ----
        v16bf vf[2][4];
#pragma unroll
        for (int kc = 0; kc < 2; ++kc)
#pragma unroll
            for (int ni = 0; ni < 4; ++ni)
#pragma unroll
                for (int i = 0; i < 16; ++i)
                    vf[kc][ni][i] =
                        vt[(kc * 32 + hi * 16 + i) * 64 + ni * 16 + l];
        // ---- O += P * V ----
#pragma unroll
        for (int kc = 0; kc < 2; ++kc)
#pragma unroll
            for (int ni = 0; ni < 4; ++ni)
                oacc[ni] = __builtin_amdgcn_wmma_f32_16x16x32_bf16(
                    false, pA[kc], false, vf[kc][ni], (short)0, oacc[ni],
                    false, false);
        __syncthreads();   // all waves done with vt/pbuf before next TDM
    }

    // ---- normalize & store (bf16) ----
#pragma unroll
    for (int ni = 0; ni < 4; ++ni)
#pragma unroll
        for (int r = 0; r < 8; ++r) {
            const int row = qbase + hi * 8 + r;
            const int col = h * DK_ + ni * 16 + l;
            ob[(rowoff + row) * D_ + col] = (__bf16)(oacc[ni][r] / lsum[r]);
        }
}

// ---------------------------------------------------------------------------
// LayerNorm( a + b ) * g + beta  — one wave per row of 512
// ---------------------------------------------------------------------------
__global__ __launch_bounds__(256)
void ln_res_kernel(const float* __restrict__ a, const float* __restrict__ bsrc,
                   const float* __restrict__ g, const float* __restrict__ be,
                   float* __restrict__ outF, __bf16* __restrict__ outB) {
    const int row  = blockIdx.x * 8 + (threadIdx.x >> 5);
    const int lane = threadIdx.x & 31;
    const float* pa = a + (size_t)row * D_;
    const float* pb = bsrc + (size_t)row * D_;
    float v[16];
    float s = 0.0f;
#pragma unroll
    for (int i = 0; i < 16; ++i) {
        v[i] = pa[lane + i * 32] + pb[lane + i * 32];
        s += v[i];
    }
#pragma unroll
    for (int msk = 1; msk < 32; msk <<= 1) s += __shfl_xor(s, msk, 32);
    const float mu = s * (1.0f / D_);
    float vs = 0.0f;
#pragma unroll
    for (int i = 0; i < 16; ++i) { const float d = v[i] - mu; vs += d * d; }
#pragma unroll
    for (int msk = 1; msk < 32; msk <<= 1) vs += __shfl_xor(vs, msk, 32);
    const float rs = rsqrtf(vs * (1.0f / D_) + 1e-5f);
#pragma unroll
    for (int i = 0; i < 16; ++i) {
        const int c = lane + i * 32;
        const float y = (v[i] - mu) * rs * g[c] + be[c];
        if (outF) outF[(size_t)row * D_ + c] = y;
        if (outB) outB[(size_t)row * D_ + c] = (__bf16)y;
    }
}

// temporal = LN(x1 + fused, ltg, ltb);  x2 = LN(x1 + temporal, n2g, n2b)
__global__ __launch_bounds__(256)
void ln_double_kernel(const float* __restrict__ x1, const float* __restrict__ fused,
                      const float* __restrict__ ltg, const float* __restrict__ ltb,
                      const float* __restrict__ n2g, const float* __restrict__ n2b,
                      float* __restrict__ x2F, __bf16* __restrict__ x2B) {
    const int row  = blockIdx.x * 8 + (threadIdx.x >> 5);
    const int lane = threadIdx.x & 31;
    const float* px = x1 + (size_t)row * D_;
    const float* pf = fused + (size_t)row * D_;
    float xv[16], v[16];
    float s = 0.0f;
#pragma unroll
    for (int i = 0; i < 16; ++i) {
        xv[i] = px[lane + i * 32];
        v[i]  = xv[i] + pf[lane + i * 32];
        s += v[i];
    }
#pragma unroll
    for (int msk = 1; msk < 32; msk <<= 1) s += __shfl_xor(s, msk, 32);
    float mu = s * (1.0f / D_);
    float vs = 0.0f;
#pragma unroll
    for (int i = 0; i < 16; ++i) { const float d = v[i] - mu; vs += d * d; }
#pragma unroll
    for (int msk = 1; msk < 32; msk <<= 1) vs += __shfl_xor(vs, msk, 32);
    float rs = rsqrtf(vs * (1.0f / D_) + 1e-5f);
    float s2 = 0.0f;
#pragma unroll
    for (int i = 0; i < 16; ++i) {
        const int c = lane + i * 32;
        const float t = (v[i] - mu) * rs * ltg[c] + ltb[c];
        v[i] = xv[i] + t;
        s2 += v[i];
    }
#pragma unroll
    for (int msk = 1; msk < 32; msk <<= 1) s2 += __shfl_xor(s2, msk, 32);
    mu = s2 * (1.0f / D_);
    vs = 0.0f;
#pragma unroll
    for (int i = 0; i < 16; ++i) { const float d = v[i] - mu; vs += d * d; }
#pragma unroll
    for (int msk = 1; msk < 32; msk <<= 1) vs += __shfl_xor(vs, msk, 32);
    rs = rsqrtf(vs * (1.0f / D_) + 1e-5f);
#pragma unroll
    for (int i = 0; i < 16; ++i) {
        const int c = lane + i * 32;
        const float y = (v[i] - mu) * rs * n2g[c] + n2b[c];
        x2F[(size_t)row * D_ + c] = y;
        x2B[(size_t)row * D_ + c] = (__bf16)y;
    }
}

// ---------------------------------------------------------------------------
// Strided conv1d (VALID, stride==kernel)
// ---------------------------------------------------------------------------
__global__ __launch_bounds__(256)
void conv1d_kernel(const float* __restrict__ x1, const float* __restrict__ w,
                   const float* __restrict__ bias, float* __restrict__ y,
                   int sc, int T) {
    const int idx = blockIdx.x * 256 + threadIdx.x;
    if (idx >= B_ * C_ * T) return;
    const int t = idx % T;
    const int c = (idx / T) % C_;
    const int b = idx / (T * C_);
    const float* xrow = x1 + ((size_t)b * S_ + (size_t)t * sc) * D_;
    const float* wc   = w + (size_t)c * D_ * sc;
    float acc = 0.0f;
    for (int d = 0; d < D_; ++d) {
        const float* xr = xrow + d;
        const float* wr = wc + (size_t)d * sc;
        for (int kk = 0; kk < sc; ++kk)
            acc += xr[(size_t)kk * D_] * wr[kk];
    }
    y[idx] = acc + bias[c];
}

// Linear interp back to S + channel concat into bf16 [M, D]
__global__ __launch_bounds__(256)
void interp_concat_kernel(const float* __restrict__ y0, const float* __restrict__ y1,
                          const float* __restrict__ y2, const float* __restrict__ y3,
                          __bf16* __restrict__ msb) {
    const int idx = blockIdx.x * 256 + threadIdx.x;
    if (idx >= M_ * D_) return;
    const int c = idx % C_;
    const int i = (idx / C_) & 3;
    const int s = (idx / D_) % S_;
    const int b = idx / (D_ * S_);
    const float* y; int T;
    if      (i == 0) { y = y0; T = 1024; }
    else if (i == 1) { y = y1; T = 256;  }
    else if (i == 2) { y = y2; T = 64;   }
    else             { y = y3; T = 16;   }
    y += ((size_t)b * C_ + c) * T;
    float val;
    if (T == S_) {
        val = y[s];
    } else {
        float pos = (s + 0.5f) * ((float)T / (float)S_) - 0.5f;
        pos = fminf(fmaxf(pos, 0.0f), (float)(T - 1));
        const int lo  = (int)floorf(pos);
        const int hiI = min(lo + 1, T - 1);
        const float w = pos - (float)lo;
        val = y[lo] * (1.0f - w) + y[hiI] * w;
    }
    msb[idx] = (__bf16)val;
}

// f32 -> bf16
__global__ __launch_bounds__(256)
void f2bf_kernel(const float* __restrict__ in, __bf16* __restrict__ out, int n) {
    const int i = blockIdx.x * 256 + threadIdx.x;
    if (i < n) out[i] = (__bf16)in[i];
}

// ---------------------------------------------------------------------------
// Host-side orchestration
// ---------------------------------------------------------------------------
extern "C" void kernel_launch(void* const* d_in, const int* in_sizes, int n_in,
                              void* d_out, int out_size, void* d_ws, size_t ws_size,
                              hipStream_t stream) {
    (void)in_sizes; (void)n_in; (void)out_size; (void)ws_size;

    const float* x    = (const float*)d_in[0];
    const int*   rm   = (const int*)d_in[1];
    const int*   am   = (const int*)d_in[2];
    const float* wq   = (const float*)d_in[3];
    const float* bq   = (const float*)d_in[4];
    const float* wk   = (const float*)d_in[5];
    const float* bk   = (const float*)d_in[6];
    const float* wv   = (const float*)d_in[7];
    const float* bv   = (const float*)d_in[8];
    const float* wo   = (const float*)d_in[9];
    const float* bo   = (const float*)d_in[10];
    const float* rbias= (const float*)d_in[11];
    const float* cw0  = (const float*)d_in[12];
    const float* cb0  = (const float*)d_in[13];
    const float* cw1  = (const float*)d_in[14];
    const float* cb1  = (const float*)d_in[15];
    const float* cw2  = (const float*)d_in[16];
    const float* cb2  = (const float*)d_in[17];
    const float* cw3  = (const float*)d_in[18];
    const float* cb3  = (const float*)d_in[19];
    const float* fw   = (const float*)d_in[20];
    const float* fb   = (const float*)d_in[21];
    const float* ltg  = (const float*)d_in[22];
    const float* ltb  = (const float*)d_in[23];
    const float* n1g  = (const float*)d_in[24];
    const float* n1b  = (const float*)d_in[25];
    const float* n2g  = (const float*)d_in[26];
    const float* n2b  = (const float*)d_in[27];
    const float* fw1  = (const float*)d_in[28];
    const float* fb1  = (const float*)d_in[29];
    const float* fw2  = (const float*)d_in[30];
    const float* fb2  = (const float*)d_in[31];
    const float* n3g  = (const float*)d_in[32];
    const float* n3b  = (const float*)d_in[33];
    float* out = (float*)d_out;

    // ---- workspace bump allocator (256B aligned) ----
    char* ws = (char*)d_ws;
    size_t off = 0;
    auto alloc = [&](size_t bytes) -> char* {
        char* p = ws + off;
        off = (off + bytes + 255) & ~(size_t)255;
        return p;
    };
    __bf16* xb   = (__bf16*)alloc((size_t)M_ * D_ * 2);
    __bf16* wqb  = (__bf16*)alloc((size_t)D_ * D_ * 2);
    __bf16* wkb  = (__bf16*)alloc((size_t)D_ * D_ * 2);
    __bf16* wvb  = (__bf16*)alloc((size_t)D_ * D_ * 2);
    __bf16* wob  = (__bf16*)alloc((size_t)D_ * D_ * 2);
    __bf16* fwb  = (__bf16*)alloc((size_t)D_ * D_ * 2);
    __bf16* fw1b = (__bf16*)alloc((size_t)FF_ * D_ * 2);
    __bf16* fw2b = (__bf16*)alloc((size_t)D_ * FF_ * 2);
    __bf16* qbuf = (__bf16*)alloc((size_t)M_ * D_ * 2);
    __bf16* kbuf = (__bf16*)alloc((size_t)M_ * D_ * 2);
    __bf16* vbuf = (__bf16*)alloc((size_t)M_ * D_ * 2);
    __bf16* obuf = (__bf16*)alloc((size_t)M_ * D_ * 2);
    float*  tmpF = (float*)alloc((size_t)M_ * D_ * 4);
    float*  x1   = (float*)alloc((size_t)M_ * D_ * 4);
    __bf16* x1b  = (__bf16*)alloc((size_t)M_ * D_ * 2);
    float*  y0   = (float*)alloc((size_t)B_ * C_ * 1024 * 4);
    float*  y1   = (float*)alloc((size_t)B_ * C_ * 256 * 4);
    float*  y2   = (float*)alloc((size_t)B_ * C_ * 64 * 4);
    float*  y3   = (float*)alloc((size_t)B_ * C_ * 16 * 4);
    __bf16* msb  = (__bf16*)alloc((size_t)M_ * D_ * 2);
    float*  x2   = (float*)alloc((size_t)M_ * D_ * 4);
    __bf16* x2b  = (__bf16*)alloc((size_t)M_ * D_ * 2);
    __bf16* hb   = (__bf16*)alloc((size_t)M_ * FF_ * 2);

    auto cvt = [&](const float* src, __bf16* dst, int n) {
        f2bf_kernel<<<(n + 255) / 256, 256, 0, stream>>>(src, dst, n);
    };
    cvt(x,   xb,   M_ * D_);
    cvt(wq,  wqb,  D_ * D_);
    cvt(wk,  wkb,  D_ * D_);
    cvt(wv,  wvb,  D_ * D_);
    cvt(wo,  wob,  D_ * D_);
    cvt(fw,  fwb,  D_ * D_);
    cvt(fw1, fw1b, FF_ * D_);
    cvt(fw2, fw2b, D_ * FF_);

    auto gemm = [&](const __bf16* A, const __bf16* W, const float* bias,
                    float* oF, __bf16* oB, int N, int K, bool relu) {
        const int blocks = (M_ / 64) * (N / 64) / 8;
        if (relu)
            gemm_bf16_kernel<1><<<blocks, 256, 0, stream>>>(A, W, bias, oF, oB, M_, N, K);
        else
            gemm_bf16_kernel<0><<<blocks, 256, 0, stream>>>(A, W, bias, oF, oB, M_, N, K);
    };

    // ---- QKV projections ----
    gemm(xb, wqb, bq, nullptr, qbuf, D_, D_, false);
    gemm(xb, wkb, bk, nullptr, kbuf, D_, D_, false);
    gemm(xb, wvb, bv, nullptr, vbuf, D_, D_, false);

    // ---- attention (double-buffered TDM V tiles) ----
    flash_attn_kernel<<<B_ * H_ * (S_ / 64), 128, ATT_SMEM_BYTES, stream>>>(
        qbuf, kbuf, vbuf, rbias, rm, am, obuf);

    // ---- output projection + LN1 ----
    gemm(obuf, wob, bo, tmpF, nullptr, D_, D_, false);
    ln_res_kernel<<<M_ / 8, 256, 0, stream>>>(x, tmpF, n1g, n1b, x1, x1b);

    // ---- multi-scale conv branch ----
    conv1d_kernel<<<(B_ * C_ * 1024 + 255) / 256, 256, 0, stream>>>(x1, cw0, cb0, y0, 1, 1024);
    conv1d_kernel<<<(B_ * C_ * 256  + 255) / 256, 256, 0, stream>>>(x1, cw1, cb1, y1, 4, 256);
    conv1d_kernel<<<(B_ * C_ * 64   + 255) / 256, 256, 0, stream>>>(x1, cw2, cb2, y2, 16, 64);
    conv1d_kernel<<<(B_ * C_ * 16   + 255) / 256, 256, 0, stream>>>(x1, cw3, cb3, y3, 64, 16);
    interp_concat_kernel<<<(M_ * D_ + 255) / 256, 256, 0, stream>>>(y0, y1, y2, y3, msb);

    // ---- fusion projection + double LN ----
    gemm(msb, fwb, fb, tmpF, nullptr, D_, D_, false);
    ln_double_kernel<<<M_ / 8, 256, 0, stream>>>(x1, tmpF, ltg, ltb, n2g, n2b, x2, x2b);

    // ---- FFN ----
    gemm(x2b, fw1b, fb1, nullptr, hb, FF_, D_, true);   // ReLU
    gemm(hb,  fw2b, fb2, tmpF, nullptr, D_, FF_, false);

    // ---- final LN -> output ----
    ln_res_kernel<<<M_ / 8, 256, 0, stream>>>(x2, tmpF, n3g, n3b, out, nullptr);
}
